// LIIF_13572096656006
// MI455X (gfx1250) — matrix operand order repaired and checked by reading, not compiled, not run on hardware
//
#include <hip/hip_runtime.h>

// ---------------------------------------------------------------------------
// Problem constants (match reference)
// ---------------------------------------------------------------------------
#define BB   8
#define HH   48
#define WW   48
#define QQ   4096
#define CC   64
#define C9   576        // C*9
#define HID  256
#define M1   (BB*QQ*4)  // 131072 rows for k/v MLP GEMMs
#define M2   (BB*QQ)    // 32768 rows for q MLP GEMMs
#define KP1  608        // 578 padded up to multiple of 32 (19*32)
#define KT1  19
#define KT8  8          // K=256 -> 8 k-tiles
#define KT18 18         // K=576 -> 18 k-tiles

typedef __attribute__((ext_vector_type(16))) __bf16 v16bf;
typedef __attribute__((ext_vector_type(8)))  __bf16 v8bf;
typedef __attribute__((ext_vector_type(8)))  float  v8f;
typedef __attribute__((ext_vector_type(4)))  int    v4i;

#if defined(__HIP_DEVICE_COMPILE__) && __has_builtin(__builtin_amdgcn_global_load_async_to_lds_b128)
#define HAS_ASYNC_LDS 1
#else
#define HAS_ASYNC_LDS 0
#endif

// ---------------------------------------------------------------------------
// bf16 helpers (manual RNE; bf16 == top 16 bits of fp32)
// ---------------------------------------------------------------------------
__device__ __forceinline__ unsigned short f2bf(float f) {
    unsigned int u = __float_as_uint(f);
    u += 0x7FFFu + ((u >> 16) & 1u);
    return (unsigned short)(u >> 16);
}
__device__ __forceinline__ float bf2f(unsigned short h) {
    return __uint_as_float(((unsigned int)h) << 16);
}

// 16-byte global -> LDS copy: async DMA on gfx1250 toolchains that expose it,
// sync load+ds_store otherwise. Builtin signature (from compiler diagnostic):
// first param = int4* in global (addrspace 1), second = LDS dest, then
// imm offset and imm cpol.
__device__ __forceinline__ void copy16_g2l(const void* g, void* l) {
#if HAS_ASYNC_LDS
    __builtin_amdgcn_global_load_async_to_lds_b128(
        (__attribute__((address_space(1))) v4i*)g,
        (__attribute__((address_space(3))) v4i*)l, 0, 0);
#else
    *(uint4*)l = *(const uint4*)g;
#endif
}

template<int N> __device__ __forceinline__ void wait_async_c() {
#if HAS_ASYNC_LDS
#if __has_builtin(__builtin_amdgcn_s_wait_asynccnt)
    __builtin_amdgcn_s_wait_asynccnt(N);
#else
    asm volatile("s_wait_asynccnt %0" :: "n"(N) : "memory");
#endif
#endif
}

// ---------------------------------------------------------------------------
// Kernel 1: 3x3 conv encoder, 3 -> 64 channels, SAME padding, fp32
// ---------------------------------------------------------------------------
__global__ void conv3x3_kernel(const float* __restrict__ inp,
                               const float* __restrict__ w,
                               const float* __restrict__ bias,
                               float* __restrict__ feat) {
    int gid = blockIdx.x * blockDim.x + threadIdx.x;
    const int total = BB * CC * HH * WW;
    if (gid >= total) return;
    int x = gid % WW;
    int y = (gid / WW) % HH;
    int o = (gid / (WW * HH)) % CC;
    int b = gid / (WW * HH * CC);
    float acc = bias[o];
    #pragma unroll
    for (int i = 0; i < 3; ++i)
        #pragma unroll
        for (int ky = 0; ky < 3; ++ky)
            #pragma unroll
            for (int kx = 0; kx < 3; ++kx) {
                int yy = y + ky - 1, xx = x + kx - 1;
                if (yy >= 0 && yy < HH && xx >= 0 && xx < WW)
                    acc += inp[((b * 3 + i) * HH + yy) * WW + xx] *
                           w[((o * 3 + i) * 3 + ky) * 3 + kx];
            }
    feat[gid] = acc;
}

// ---------------------------------------------------------------------------
// Kernel 2: pack fp32 weight W[K x N] into WMMA B tiles (bf16).
// Tile (kt,nt) = 1024B: lo half = elements j=0..7 (K=k0+(l>>4)*8+j) laid out
// as lane*16B, hi half (j=8..15) at +512B. This is both the DMA source layout
// and the LDS layout, so staging is a straight byte copy.
// ---------------------------------------------------------------------------
__global__ void pack_w_kernel(const float* __restrict__ W, int K, int N,
                              int Kt, int Nt, unsigned short* __restrict__ dst) {
    int gid = blockIdx.x * blockDim.x + threadIdx.x;
    int total = Kt * Nt * 512;
    if (gid >= total) return;
    int j    = gid & 15;
    int lane = (gid >> 4) & 31;
    int tile = gid >> 9;
    int kt = tile % Kt;
    int nt = tile / Kt;
    int k = kt * 32 + (j >> 3) * 16 + (lane >> 4) * 8 + (j & 7);
    int n = nt * 16 + (lane & 15);
    float v = (k < K && n < N) ? W[(size_t)k * N + n] : 0.0f;
    dst[(size_t)tile * 512 + (j >> 3) * 256 + lane * 8 + (j & 7)] = f2bf(v);
}

// ---------------------------------------------------------------------------
// nearest-index helper (jnp.round == round-half-even via rintf)
// ---------------------------------------------------------------------------
__device__ __forceinline__ int nearest_i(float c, int n) {
    float r = rintf(((c + 1.0f) * (float)n - 1.0f) * 0.5f);
    r = fminf(fmaxf(r, 0.0f), (float)(n - 1));
    return (int)r;
}

// unfold gather: fu[b, idx, c9] with zero padding ring
__device__ __forceinline__ float fu_gather(const float* __restrict__ feat,
                                           int b, int col, int iy, int ix) {
    int c = col / 9, p = col % 9;
    int yy = iy + p / 3 - 1, xx = ix + p % 3 - 1;
    if (yy < 0 || yy >= HH || xx < 0 || xx >= WW) return 0.0f;
    return feat[(((size_t)b * CC + c) * HH + yy) * WW + xx];
}

// ---------------------------------------------------------------------------
// Kernel 3: build ink matrix X [M1 x KP1] bf16.
// Row r = (b*Q+q)*4 + s; cols: [0,576)=kv gather, 576=rel_y, 577=rel_x, rest 0
// ---------------------------------------------------------------------------
__global__ void build_x_kernel(const float* __restrict__ feat,
                               const float* __restrict__ coord,
                               const float* __restrict__ cell,
                               unsigned short* __restrict__ X) {
    long gid = (long)blockIdx.x * blockDim.x + threadIdx.x;
    const long total = (long)M1 * KP1;
    if (gid >= total) return;
    int col = (int)(gid % KP1);
    int r   = (int)(gid / KP1);
    int s  = r & 3;
    int bq = r >> 2;
    int b  = bq >> 12;                   // Q = 4096
    float c0 = coord[(size_t)bq * 2 + 0];
    float c1 = coord[(size_t)bq * 2 + 1];
    // reference uses cell of q==0 only: tx = (H-1)/(1-cell[:,0,0])
    float rx = (1.0f - cell[(size_t)b * QQ * 2 + 0]) * (1.0f / (HH - 1));
    float ry = (1.0f - cell[(size_t)b * QQ * 2 + 1]) * (1.0f / (WW - 1));
    float sx = (s & 2) ? 1.0f : -1.0f;   // vx outer loop
    float sy = (s & 1) ? 1.0f : -1.0f;   // vy inner loop
    float py = fminf(fmaxf(c0 + sx * rx + 1e-6f, -1.0f + 1e-6f), 1.0f - 1e-6f);
    float px = fminf(fmaxf(c1 + sy * ry + 1e-6f, -1.0f + 1e-6f), 1.0f - 1e-6f);
    int iy = nearest_i(py, HH);
    int ix = nearest_i(px, WW);
    float val;
    if (col < C9) {
        val = fu_gather(feat, b, col, iy, ix);
    } else if (col == C9) {
        val = (c0 - (-1.0f + (2.0f * iy + 1.0f) / HH)) * (float)HH;
    } else if (col == C9 + 1) {
        val = (c1 - (-1.0f + (2.0f * ix + 1.0f) / WW)) * (float)WW;
    } else {
        val = 0.0f;
    }
    X[gid] = f2bf(val);
}

// ---------------------------------------------------------------------------
// Kernel 4 (workhorse): bf16 WMMA GEMM, OUT = f(A @ W + bias)
//   block = 256 threads = 8 waves (4 m-waves x 2 n-waves)
//   block tile 128(M) x 64(N); wave tile 32(M) x 32(N) -> 4 WMMA / k-step
//   A and B staged to LDS via async global->LDS DMA, double buffered.
//   LDS: A rows padded to 80B (bank-conflict-free), B tiles in lo/hi halves.
// ---------------------------------------------------------------------------
#define A_PITCH_B 80
#define A_BYTES   (128 * A_PITCH_B)       // 10240
#define B_BYTES   4096                    // 4 tiles * 1024B
#define BUF_BYTES (A_BYTES + B_BYTES)     // 14336

template<bool RELU, bool MUL, bool OUTF32>
__global__ __launch_bounds__(256)
void gemm_wmma_kernel(const unsigned short* __restrict__ A, int pitchA,
                      const unsigned short* __restrict__ Bp, int Kt, int Ntiles,
                      const float* __restrict__ bias, int Nout,
                      const unsigned short* __restrict__ mulsrc, int pitchMul,
                      unsigned short* __restrict__ outBf,
                      float* __restrict__ outF, int pitchOut) {
    __shared__ unsigned char smem[2 * BUF_BYTES];

    const int tid = threadIdx.x;
    const int l   = tid & 31;
    const int wid = tid >> 5;
    const int wm  = wid & 3;          // 4 wave rows (32 M each)
    const int wn  = wid >> 2;         // 2 wave cols (32 N each)
    const int lane16 = l & 15;
    const int half   = l >> 4;
    const int m_blk  = blockIdx.x * 128;

    v8f acc[2][2];
    acc[0][0] = (v8f){0.f,0.f,0.f,0.f,0.f,0.f,0.f,0.f};
    acc[0][1] = acc[0][0]; acc[1][0] = acc[0][0]; acc[1][1] = acc[0][0];

    // ---- stage one 32-wide k-slice (A: 128x32, B: 4 tiles) into buffer kt&1
    auto stage = [&](int kt) {
        unsigned char* buf = smem + (kt & 1) * BUF_BYTES;
        #pragma unroll
        for (int i = 0; i < 2; ++i) {                 // A: 512 x 16B chunks
            int c   = tid + i * 256;
            int row = c >> 2, col = c & 3;
            const unsigned char* g = (const unsigned char*)A +
                (size_t)(m_blk + row) * pitchA * 2 + (size_t)kt * 64 + col * 16;
            copy16_g2l(g, buf + row * A_PITCH_B + col * 16);
        }
        {                                             // B: 256 x 16B chunks
            int nt_local = tid >> 6;                  // 64 chunks per tile
            int co       = tid & 63;
            int ntg = blockIdx.y * 4 + nt_local;
            if (ntg >= Ntiles) ntg = Ntiles - 1;      // clamp (stores guarded)
            const unsigned char* g = (const unsigned char*)Bp +
                ((size_t)ntg * Kt + kt) * 1024 + co * 16;
            copy16_g2l(g, buf + A_BYTES + nt_local * 1024 + co * 16);
        }
    };

    // ---- consume one staged k-slice: 4 WMMAs
    auto consume = [&](int kt) {
        const unsigned char* buf = smem + (kt & 1) * BUF_BYTES;
        const unsigned char* ab = buf + (wm * 32 + lane16) * A_PITCH_B + half * 16;
        v8bf a0lo = *(const v8bf*)(ab);
        v8bf a0hi = *(const v8bf*)(ab + 32);
        v8bf a1lo = *(const v8bf*)(ab + 16 * A_PITCH_B);
        v8bf a1hi = *(const v8bf*)(ab + 16 * A_PITCH_B + 32);
        v16bf a0 = __builtin_shufflevector(a0lo, a0hi, 0,1,2,3,4,5,6,7,8,9,10,11,12,13,14,15);
        v16bf a1 = __builtin_shufflevector(a1lo, a1hi, 0,1,2,3,4,5,6,7,8,9,10,11,12,13,14,15);
        const unsigned char* bb = buf + A_BYTES + (wn * 2) * 1024 + l * 16;
        v8bf b0lo = *(const v8bf*)(bb);
        v8bf b0hi = *(const v8bf*)(bb + 512);
        v8bf b1lo = *(const v8bf*)(bb + 1024);
        v8bf b1hi = *(const v8bf*)(bb + 1536);
        v16bf b0 = __builtin_shufflevector(b0lo, b0hi, 0,1,2,3,4,5,6,7,8,9,10,11,12,13,14,15);
        v16bf b1 = __builtin_shufflevector(b1lo, b1hi, 0,1,2,3,4,5,6,7,8,9,10,11,12,13,14,15);
        acc[0][0] = __builtin_amdgcn_wmma_f32_16x16x32_bf16(false, a0, false, b0, (short)0, acc[0][0], false, false);
        acc[1][0] = __builtin_amdgcn_wmma_f32_16x16x32_bf16(false, a1, false, b0, (short)0, acc[1][0], false, false);
        acc[0][1] = __builtin_amdgcn_wmma_f32_16x16x32_bf16(false, a0, false, b1, (short)0, acc[0][1], false, false);
        acc[1][1] = __builtin_amdgcn_wmma_f32_16x16x32_bf16(false, a1, false, b1, (short)0, acc[1][1], false, false);
    };

    stage(0);
    for (int kt = 0; kt < Kt - 1; ++kt) {
        stage(kt + 1);            // DMA next slice into the other buffer
        wait_async_c<3>();        // this wave's loads for slice kt are done
        __syncthreads();          // everyone's slice-kt data visible
        consume(kt);
        __syncthreads();          // reads done before slice kt+2 overwrites
    }
    wait_async_c<0>();
    __syncthreads();
    consume(Kt - 1);

    // ---- epilogue: bias (+ReLU | *kv), store bf16 or fp32, per-lane guarded
    #pragma unroll
    for (int sn = 0; sn < 2; ++sn) {
        const int ntg = blockIdx.y * 4 + wn * 2 + sn;
        const int n = ntg * 16 + lane16;
        if (n < Nout) {
            const float bv = bias[n];
            #pragma unroll
            for (int t = 0; t < 2; ++t) {
                #pragma unroll
                for (int rr = 0; rr < 8; ++rr) {
                    const int m = m_blk + wm * 32 + t * 16 + half * 8 + rr;
                    float v = acc[t][sn][rr] + bv;
                    if (RELU) v = fmaxf(v, 0.0f);
                    if (MUL)  v *= bf2f(mulsrc[(size_t)m * pitchMul + n]);
                    if (OUTF32) outF[(size_t)m * pitchOut + n] = v;
                    else        outBf[(size_t)m * pitchOut + n] = f2bf(v);
                }
            }
        }
    }
}

// ---------------------------------------------------------------------------
// Kernel 5: attention (one wave32 per query)
// ---------------------------------------------------------------------------
__global__ __launch_bounds__(256)
void attn_kernel(const float* __restrict__ feat,
                 const float* __restrict__ coord,
                 const unsigned short* __restrict__ predk,
                 const unsigned short* __restrict__ predv,
                 unsigned short* __restrict__ X2) {
    int gwave = (blockIdx.x * blockDim.x + threadIdx.x) >> 5;
    int l = threadIdx.x & 31;
    if (gwave >= M2) return;
    int b = gwave >> 12;              // Q = 4096
    float c0 = coord[(size_t)gwave * 2 + 0];
    float c1 = coord[(size_t)gwave * 2 + 1];
    int iy = nearest_i(c0, HH);
    int ix = nearest_i(c1, WW);

    size_t rbase = (size_t)gwave * 4 * C9;
    float s0 = 0.f, s1 = 0.f, s2 = 0.f, s3 = 0.f;
    for (int c = l; c < C9; c += 32) {
        float qv = fu_gather(feat, b, c, iy, ix);
        s0 += qv * bf2f(predk[rbase + 0 * C9 + c]);
        s1 += qv * bf2f(predk[rbase + 1 * C9 + c]);
        s2 += qv * bf2f(predk[rbase + 2 * C9 + c]);
        s3 += qv * bf2f(predk[rbase + 3 * C9 + c]);
    }
    #pragma unroll
    for (int off = 16; off > 0; off >>= 1) {
        s0 += __shfl_xor(s0, off, 32);
        s1 += __shfl_xor(s1, off, 32);
        s2 += __shfl_xor(s2, off, 32);
        s3 += __shfl_xor(s3, off, 32);
    }
    float mx = fmaxf(fmaxf(s0, s1), fmaxf(s2, s3));
    float w0 = __expf(s0 - mx), w1 = __expf(s1 - mx);
    float w2 = __expf(s2 - mx), w3 = __expf(s3 - mx);
    float inv = 1.0f / (w0 + w1 + w2 + w3);
    w0 *= inv; w1 *= inv; w2 *= inv; w3 *= inv;

    for (int c = l; c < C9; c += 32) {
        float x = w0 * bf2f(predv[rbase + 0 * C9 + c]) +
                  w1 * bf2f(predv[rbase + 1 * C9 + c]) +
                  w2 * bf2f(predv[rbase + 2 * C9 + c]) +
                  w3 * bf2f(predv[rbase + 3 * C9 + c]);
        X2[(size_t)gwave * C9 + c] = f2bf(x);
    }
}

// ---------------------------------------------------------------------------
// Host-side launch
// ---------------------------------------------------------------------------
static inline size_t alignup(size_t v) { return (v + 255) & ~(size_t)255; }

extern "C" void kernel_launch(void* const* d_in, const int* in_sizes, int n_in,
                              void* d_out, int out_size, void* d_ws, size_t ws_size,
                              hipStream_t stream) {
    const float* inp   = (const float*)d_in[0];
    const float* coord = (const float*)d_in[1];
    const float* cell  = (const float*)d_in[2];
    const float* enc_w = (const float*)d_in[3];
    const float* enc_b = (const float*)d_in[4];
    const float* kW1 = (const float*)d_in[5];  const float* kb1 = (const float*)d_in[6];
    const float* kW2 = (const float*)d_in[7];  const float* kb2 = (const float*)d_in[8];
    const float* kW3 = (const float*)d_in[9];  const float* kb3 = (const float*)d_in[10];
    const float* vW1 = (const float*)d_in[11]; const float* vb1 = (const float*)d_in[12];
    const float* vW2 = (const float*)d_in[13]; const float* vb2 = (const float*)d_in[14];
    const float* vW3 = (const float*)d_in[15]; const float* vb3 = (const float*)d_in[16];
    const float* qW1 = (const float*)d_in[17]; const float* qb1 = (const float*)d_in[18];
    const float* qW2 = (const float*)d_in[19]; const float* qb2 = (const float*)d_in[20];
    const float* qW3 = (const float*)d_in[21]; const float* qb3 = (const float*)d_in[22];
    float* outF = (float*)d_out;

    // ---- carve workspace (256B aligned) ----
    char* p = (char*)d_ws;
    float* feat = (float*)p;            p += alignup((size_t)BB * CC * HH * WW * 4);
    unsigned short* kW1p = (unsigned short*)p; p += alignup((size_t)KT1  * 16 * 512 * 2);
    unsigned short* kW2p = (unsigned short*)p; p += alignup((size_t)KT8  * 16 * 512 * 2);
    unsigned short* kW3p = (unsigned short*)p; p += alignup((size_t)KT8  * 36 * 512 * 2);
    unsigned short* vW1p = (unsigned short*)p; p += alignup((size_t)KT1  * 16 * 512 * 2);
    unsigned short* vW2p = (unsigned short*)p; p += alignup((size_t)KT8  * 16 * 512 * 2);
    unsigned short* vW3p = (unsigned short*)p; p += alignup((size_t)KT8  * 36 * 512 * 2);
    unsigned short* qW1p = (unsigned short*)p; p += alignup((size_t)KT18 * 16 * 512 * 2);
    unsigned short* qW2p = (unsigned short*)p; p += alignup((size_t)KT8  * 16 * 512 * 2);
    unsigned short* qW3p = (unsigned short*)p; p += alignup((size_t)KT8  *  1 * 512 * 2);
    unsigned short* X    = (unsigned short*)p; p += alignup((size_t)M1 * KP1 * 2);
    unsigned short* H1   = (unsigned short*)p; p += alignup((size_t)M1 * HID * 2);
    unsigned short* H2   = (unsigned short*)p; p += alignup((size_t)M1 * HID * 2);
    unsigned short* pk   = (unsigned short*)p; p += alignup((size_t)M1 * C9 * 2);
    unsigned short* pv   = (unsigned short*)p; p += alignup((size_t)M1 * C9 * 2);
    unsigned short* X2   = (unsigned short*)p; p += alignup((size_t)M2 * C9 * 2);
    (void)ws_size; (void)n_in; (void)in_sizes; (void)out_size;

    // ---- 1. encoder conv ----
    {
        int total = BB * CC * HH * WW;
        conv3x3_kernel<<<(total + 255) / 256, 256, 0, stream>>>(inp, enc_w, enc_b, feat);
    }

    // ---- 2. pack weights into WMMA B-tiles ----
    auto pack = [&](const float* Wm, int K, int N, int Kt, int Nt, unsigned short* dst) {
        int total = Kt * Nt * 512;
        pack_w_kernel<<<(total + 255) / 256, 256, 0, stream>>>(Wm, K, N, Kt, Nt, dst);
    };
    pack(kW1, C9 + 2, HID, KT1, 16, kW1p);
    pack(kW2, HID,    HID, KT8, 16, kW2p);
    pack(kW3, HID,    C9,  KT8, 36, kW3p);
    pack(vW1, C9 + 2, HID, KT1, 16, vW1p);
    pack(vW2, HID,    HID, KT8, 16, vW2p);
    pack(vW3, HID,    C9,  KT8, 36, vW3p);
    pack(qW1, C9,     HID, KT18, 16, qW1p);
    pack(qW2, HID,    HID, KT8,  16, qW2p);
    pack(qW3, HID,    3,   KT8,   1, qW3p);

    // ---- 3. build ink matrix X ----
    {
        long total = (long)M1 * KP1;
        build_x_kernel<<<(int)((total + 255) / 256), 256, 0, stream>>>(feat, coord, cell, X);
    }

    // ---- 4. k-MLP and v-MLP (three WMMA GEMMs each, fused epilogues) ----
    dim3 blk(256);
    // k: 578->256 relu -> 256->256 relu -> 256->576, *kv
    gemm_wmma_kernel<true,  false, false><<<dim3(M1/128, 4), blk, 0, stream>>>(
        X,  KP1, kW1p, KT1, 16, kb1, HID, nullptr, 0, H1, nullptr, HID);
    gemm_wmma_kernel<true,  false, false><<<dim3(M1/128, 4), blk, 0, stream>>>(
        H1, HID, kW2p, KT8, 16, kb2, HID, nullptr, 0, H2, nullptr, HID);
    gemm_wmma_kernel<false, true,  false><<<dim3(M1/128, 9), blk, 0, stream>>>(
        H2, HID, kW3p, KT8, 36, kb3, C9,  X, KP1, pk, nullptr, C9);
    // v
    gemm_wmma_kernel<true,  false, false><<<dim3(M1/128, 4), blk, 0, stream>>>(
        X,  KP1, vW1p, KT1, 16, vb1, HID, nullptr, 0, H1, nullptr, HID);
    gemm_wmma_kernel<true,  false, false><<<dim3(M1/128, 4), blk, 0, stream>>>(
        H1, HID, vW2p, KT8, 16, vb2, HID, nullptr, 0, H2, nullptr, HID);
    gemm_wmma_kernel<false, true,  false><<<dim3(M1/128, 9), blk, 0, stream>>>(
        H2, HID, vW3p, KT8, 36, vb3, C9,  X, KP1, pv, nullptr, C9);

    // ---- 5. attention + softmax blend -> X2 ----
    attn_kernel<<<M2 / 8, 256, 0, stream>>>(feat, coord, pk, pv, X2);

    // ---- 6. q-MLP: 576->256 relu -> 256->256 relu -> 256->3 (fp32 out) ----
    gemm_wmma_kernel<true,  false, false><<<dim3(M2/128, 4), blk, 0, stream>>>(
        X2, C9,  qW1p, KT18, 16, qb1, HID, nullptr, 0, H1, nullptr, HID);
    gemm_wmma_kernel<true,  false, false><<<dim3(M2/128, 4), blk, 0, stream>>>(
        H1, HID, qW2p, KT8,  16, qb2, HID, nullptr, 0, H2, nullptr, HID);
    gemm_wmma_kernel<false, false, true ><<<dim3(M2/128, 1), blk, 0, stream>>>(
        H2, HID, qW3p, KT8,   1, qb3, 3,   nullptr, 0, nullptr, outF, 3);
}
